// HybridQLSTM_65481071400051
// MI455X (gfx1250) — matrix operand order repaired and checked by reading, not compiled, not run on hardware
//
#include <hip/hip_runtime.h>
#include <hip/hip_bf16.h>

// ---------------------------------------------------------------------------
// Hybrid QLSTM on gfx1250 (MI455X), wave32 + WMMA bf16 + TDM staging.
//
//   Kernel 1: pack Wc/Wq (fp32 [4,512,256] each) into bf16 B-fragment layout
//             for v_wmma_f32_16x16x32_bf16:
//               K = kt*32 + (L/16)*8 + e + (e>=8 ? 8 : 0) ,  N = nt*16 + (L%16)
//   Kernel 2: persistent recurrence. 32 WGs (4 batch-tiles x 8 H-tiles),
//             8 waves each; wave w computes gate w's 64x32 tile with
//             4x2 f32 accum fragments over K=512 (x part + h part).
//             h carried between steps as bf16 (double-buffered in d_ws) and
//             staged into LDS by the Tensor Data Mover (tensor_load_to_lds,
//             2D descriptor with LDS row padding); c carried in registers.
//             Device-wide sync: monotonic atomic counter barrier.
// ---------------------------------------------------------------------------

typedef __bf16 bf16_t;
typedef __attribute__((ext_vector_type(16))) __bf16 v16bf;
typedef __attribute__((ext_vector_type(8)))  __bf16 v8bf;
typedef __attribute__((ext_vector_type(8)))  float  v8f;
typedef __attribute__((ext_vector_type(4)))  unsigned u32x4;
typedef __attribute__((ext_vector_type(8)))  unsigned u32x8;

#define TT 512
#define BB 256
#define HH 256
#define DD 512
#define NBLK 32                      // persistent workgroups

#define LROW 264                     // 256 elems + 8 bf16 pad (16 B) per row

#define WPK_BYTES (8u*16u*16u*32u*16u*2u)  // 2 MiB packed bf16 weights
#define HBUF_BYTES (2u*BB*HH*2u)           // ping-pong bf16 h state

__device__ __forceinline__ unsigned short f2bfu(float f) {
  // round-to-nearest-even fp32 -> bf16
  unsigned u = __float_as_uint(f);
  unsigned r = u + 0x7FFFu + ((u >> 16) & 1u);
  return (unsigned short)(r >> 16);
}

__device__ __forceinline__ float sigf(float x) {
  return 1.0f / (1.0f + __expf(-x));
}

__device__ __forceinline__ size_t wpk_idx(int g, int kt, int nt, int lane) {
  return ((((size_t)g * 16 + (size_t)kt) * 16 + (size_t)nt) * 32 + (size_t)lane) * 16;
}

// ---------------------------------------------------------------------------
// Kernel 1: weight packing (runs every launch; cheap, deterministic)
// ---------------------------------------------------------------------------
__global__ void qlstm_pack_w(const float* __restrict__ Wc,
                             const float* __restrict__ Wq,
                             bf16_t* __restrict__ Wpk) {
  const int b    = blockIdx.x;        // 0..2047 : (g, kt, nt)
  const int lane = threadIdx.x;       // 0..31
  const int g  = b >> 8;
  const int kt = (b >> 4) & 15;
  const int nt = b & 15;
  const float* W = (g < 4) ? (Wc + (size_t)g * DD * HH)
                           : (Wq + (size_t)(g - 4) * DD * HH);
  const int Kbase = kt * 32 + (lane >> 4) * 8;
  const int N     = nt * 16 + (lane & 15);
  unsigned short tmp[16];
#pragma unroll
  for (int e = 0; e < 16; ++e)
    tmp[e] = f2bfu(W[(size_t)(Kbase + e + (e >= 8 ? 8 : 0)) * HH + N]);
  uint4* dst = (uint4*)(Wpk + wpk_idx(g, kt, nt, lane));
  uint4 lo, hi;
  lo.x = (unsigned)tmp[0]  | ((unsigned)tmp[1]  << 16);
  lo.y = (unsigned)tmp[2]  | ((unsigned)tmp[3]  << 16);
  lo.z = (unsigned)tmp[4]  | ((unsigned)tmp[5]  << 16);
  lo.w = (unsigned)tmp[6]  | ((unsigned)tmp[7]  << 16);
  hi.x = (unsigned)tmp[8]  | ((unsigned)tmp[9]  << 16);
  hi.y = (unsigned)tmp[10] | ((unsigned)tmp[11] << 16);
  hi.z = (unsigned)tmp[12] | ((unsigned)tmp[13] << 16);
  hi.w = (unsigned)tmp[14] | ((unsigned)tmp[15] << 16);
  dst[0] = lo;
  dst[1] = hi;
}

// ---------------------------------------------------------------------------
// Kernel 2: persistent recurrent kernel
// ---------------------------------------------------------------------------
__global__ __launch_bounds__(256) void qlstm_persistent(
    const float* __restrict__ x,     // [T,B,IN]
    const float* __restrict__ bc,    // [4,H]
    const float* __restrict__ bq,    // [4,H]
    const float* __restrict__ qs,    // [1]
    const bf16_t* __restrict__ Wpk,  // packed weights
    bf16_t* __restrict__ hbuf,       // [2][B][H] bf16 ping-pong
    unsigned* __restrict__ bar,      // grid barrier counter (zeroed per launch)
    float* __restrict__ out)         // stacked [T,B,H] then hT, cT
{
  // LDS: x half (K=0..255) and h half (K=256..511) of the activation tile,
  //      each 64 rows x 264 bf16 (8-elem pad -> 528 B row stride, bank-clean,
  //      and TDM-pad compatible: 512 B data + 16 B pad per row).
  __shared__ bf16_t xstage[64][LROW];
  __shared__ bf16_t hstage[64][LROW];
  __shared__ float  xch[8][64][33];

  const int tid  = threadIdx.x;
  const int wave = tid >> 5;          // gate index 0..7 (0-3: Wc, 4-7: Wq)
  const int lane = tid & 31;
  const int halfSel = lane >> 4;
  const int lm   = lane & 15;
  const int bt   = blockIdx.x >> 3;   // batch tile 0..3  (64 rows)
  const int ntg  = blockIdx.x & 7;    // H tile 0..7      (32 cols)

  // staging mapping: 4 threads per row, 64 cols each
  const int srow = tid >> 2;
  const int scol = (tid & 3) * 64;
  // elementwise mapping: row em, 8 consecutive cols from en0
  const int em  = tid >> 2;
  const int en0 = (tid & 3) * 8;

  const float q = qs[0];
  const float omq = 1.0f - q;

  float bcv[4][8], bqv[4][8];
#pragma unroll
  for (int g = 0; g < 4; ++g)
#pragma unroll
    for (int j = 0; j < 8; ++j) {
      const int ncol = ntg * 32 + en0 + j;
      bcv[g][j] = bc[g * HH + ncol];
      bqv[g][j] = bq[g * HH + ncol];
    }

  float creg[8];
#pragma unroll
  for (int j = 0; j < 8; ++j) creg[j] = 0.0f;

  for (int t = 0; t < TT; ++t) {
    const bf16_t* h_in  = hbuf + (size_t)(t & 1) * (BB * HH);
    bf16_t*       h_out = hbuf + (size_t)((t + 1) & 1) * (BB * HH);

    // ---- issue TDM: DMA h_{t-1} tile (64 x 256 bf16) global -> LDS ----
    // D# group0: count=1 | lds_addr | global_addr | type=2
    // D# group1: data_size=2B, pad_enable, pad_interval=128 dw, pad_amount=4 dw,
    //            tensor_dim0=256, tensor_dim1=64, tile 256x64, dim0_stride=256
    if (wave == 0) {
      unsigned long long ga =
          (unsigned long long)(uintptr_t)(h_in + (size_t)bt * 64 * HH);
      u32x4 g0;
      g0[0] = 1u;                                     // count=1, no gather
      g0[1] = (unsigned)(uintptr_t)&hstage[0][0];     // lds_addr (bytes)
      g0[2] = (unsigned)ga;                           // global_addr lo
      g0[3] = ((unsigned)(ga >> 32) & 0x01FFFFFFu)    // global_addr hi
              | 0x80000000u;                          // type=2 ("image")
      u32x8 g1;
      g1[0] = 0x07910000u;   // data_size=1(2B)<<16 | pad_en<<20 | 6<<22 | 3<<25
      g1[1] = 0x01000000u;   // tensor_dim0 = 256  (bits 79:48, low half)
      g1[2] = 0x00400000u;   // tensor_dim1 = 64   (bits 111:80, low half)
      g1[3] = 0x01000000u;   // tile_dim0   = 256  (bits 127:112)
      g1[4] = 64u;           // tile_dim1   = 64   (bits 143:128)
      g1[5] = 256u;          // tensor_dim0_stride = 256 (bits 207:160, lo)
      g1[6] = 0u;
      g1[7] = 0u;
      asm volatile("tensor_load_to_lds %0, %1"
                   :: "s"(g0), "s"(g1)
                   : "memory");
    }

    // ---- stage x_t tile (fp32 -> bf16) into LDS with VALU, overlapping TDM
    {
      const float4* xs =
          (const float4*)(x + (((size_t)t * BB) + bt * 64 + srow) * 256 + scol);
#pragma unroll
      for (int i = 0; i < 16; ++i) {
        float4 v = xs[i];
        uint2 pk;
        pk.x = (unsigned)f2bfu(v.x) | ((unsigned)f2bfu(v.y) << 16);
        pk.y = (unsigned)f2bfu(v.z) | ((unsigned)f2bfu(v.w) << 16);
        *(uint2*)&xstage[srow][scol + i * 4] = pk;
      }
    }

    // TDM completion (issuing wave), then publish LDS to all waves
    if (wave == 0) __builtin_amdgcn_s_wait_tensorcnt(0);
    __syncthreads();

    // ---- GEMM: wave w computes gate w, 64x32 tile, K = 512 ----
    v8f acc[4][2];
#pragma unroll
    for (int mt = 0; mt < 4; ++mt)
#pragma unroll
      for (int nt = 0; nt < 2; ++nt)
#pragma unroll
        for (int r = 0; r < 8; ++r) acc[mt][nt][r] = 0.0f;

    // x half: kt = 0..7 (K = 0..255)
#pragma unroll 2
    for (int kt = 0; kt < 8; ++kt) {
      __builtin_prefetch(Wpk + wpk_idx(wave, kt + 1, ntg * 2, lane), 0, 0);
      v16bf bfrg[2];
#pragma unroll
      for (int nt = 0; nt < 2; ++nt) {
        const v8bf* bp =
            (const v8bf*)(Wpk + wpk_idx(wave, kt, ntg * 2 + nt, lane));
        bfrg[nt] = __builtin_shufflevector(bp[0], bp[1],
            0, 1, 2, 3, 4, 5, 6, 7, 8, 9, 10, 11, 12, 13, 14, 15);
      }
#pragma unroll
      for (int mt = 0; mt < 4; ++mt) {
        const bf16_t* ap = &xstage[mt * 16 + lm][kt * 32 + halfSel * 8];
        v8bf alo = *(const v8bf*)ap;
        v8bf ahi = *(const v8bf*)(ap + 16);
        v16bf afrg = __builtin_shufflevector(alo, ahi,
            0, 1, 2, 3, 4, 5, 6, 7, 8, 9, 10, 11, 12, 13, 14, 15);
        acc[mt][0] = __builtin_amdgcn_wmma_f32_16x16x32_bf16(
            false, afrg, false, bfrg[0], (short)0, acc[mt][0], false, false);
        acc[mt][1] = __builtin_amdgcn_wmma_f32_16x16x32_bf16(
            false, afrg, false, bfrg[1], (short)0, acc[mt][1], false, false);
      }
    }

    // h half: kt = 8..15 (K = 256..511)
#pragma unroll 2
    for (int kh = 0; kh < 8; ++kh) {
      if (kh + 1 < 8)
        __builtin_prefetch(Wpk + wpk_idx(wave, kh + 9, ntg * 2, lane), 0, 0);
      v16bf bfrg[2];
#pragma unroll
      for (int nt = 0; nt < 2; ++nt) {
        const v8bf* bp =
            (const v8bf*)(Wpk + wpk_idx(wave, kh + 8, ntg * 2 + nt, lane));
        bfrg[nt] = __builtin_shufflevector(bp[0], bp[1],
            0, 1, 2, 3, 4, 5, 6, 7, 8, 9, 10, 11, 12, 13, 14, 15);
      }
#pragma unroll
      for (int mt = 0; mt < 4; ++mt) {
        const bf16_t* ap = &hstage[mt * 16 + lm][kh * 32 + halfSel * 8];
        v8bf alo = *(const v8bf*)ap;
        v8bf ahi = *(const v8bf*)(ap + 16);
        v16bf afrg = __builtin_shufflevector(alo, ahi,
            0, 1, 2, 3, 4, 5, 6, 7, 8, 9, 10, 11, 12, 13, 14, 15);
        acc[mt][0] = __builtin_amdgcn_wmma_f32_16x16x32_bf16(
            false, afrg, false, bfrg[0], (short)0, acc[mt][0], false, false);
        acc[mt][1] = __builtin_amdgcn_wmma_f32_16x16x32_bf16(
            false, afrg, false, bfrg[1], (short)0, acc[mt][1], false, false);
      }
    }

    // ---- scatter D fragments to LDS exchange buffer ----
    // D layout: lane L holds (M = mt*16 + (L/16)*8 + r, N = nt*16 + L%16)
#pragma unroll
    for (int mt = 0; mt < 4; ++mt)
#pragma unroll
      for (int nt = 0; nt < 2; ++nt)
#pragma unroll
        for (int r = 0; r < 8; ++r)
          xch[wave][mt * 16 + halfSel * 8 + r][nt * 16 + lm] = acc[mt][nt][r];
    __syncthreads();

    // ---- fused gate math + cell/hidden update ----
    {
      const int gb = bt * 64 + em;
#pragma unroll
      for (int j = 0; j < 8; ++j) {
        const int nn   = en0 + j;
        const int ncol = ntg * 32 + nn;
        float zcf = xch[0][em][nn] + bcv[0][j];
        float zci = xch[1][em][nn] + bcv[1][j];
        float zcg = xch[2][em][nn] + bcv[2][j];
        float zco = xch[3][em][nn] + bcv[3][j];
        float zqf = tanhf(xch[4][em][nn] + bqv[0][j]);
        float zqi = tanhf(xch[5][em][nn] + bqv[1][j]);
        float zqg = tanhf(xch[6][em][nn] + bqv[2][j]);
        float zqo = tanhf(xch[7][em][nn] + bqv[3][j]);
        float fg = omq * sigf(zcf) + q * sigf(zqf);
        float ig = omq * sigf(zci) + q * sigf(zqi);
        float gg = omq * tanhf(zcg) + q * tanhf(zqg);
        float og = omq * sigf(zco) + q * sigf(zqo);
        creg[j] = fg * creg[j] + ig * gg;
        float h = og * tanhf(creg[j]);
        ((unsigned short*)h_out)[(size_t)gb * HH + ncol] = f2bfu(h);
        out[((size_t)t * BB + gb) * HH + ncol] = h;
        if (t == TT - 1) {
          out[(size_t)TT * BB * HH + (size_t)gb * HH + ncol] = h;
          out[(size_t)TT * BB * HH + (size_t)BB * HH + (size_t)gb * HH + ncol] =
              creg[j];
        }
      }
    }

    // ---- device-wide barrier (monotonic counter, reset per launch) ----
    __threadfence();
    __syncthreads();
    if (tid == 0) {
      __hip_atomic_fetch_add(bar, 1u, __ATOMIC_ACQ_REL,
                             __HIP_MEMORY_SCOPE_AGENT);
      const unsigned tgt = (unsigned)(t + 1) * (unsigned)NBLK;
      while (__hip_atomic_load(bar, __ATOMIC_ACQUIRE,
                               __HIP_MEMORY_SCOPE_AGENT) < tgt)
        __builtin_amdgcn_s_sleep(1);
    }
    __syncthreads();
  }
}

// ---------------------------------------------------------------------------
extern "C" void kernel_launch(void* const* d_in, const int* in_sizes, int n_in,
                              void* d_out, int out_size, void* d_ws,
                              size_t ws_size, hipStream_t stream) {
  (void)in_sizes; (void)n_in; (void)out_size; (void)ws_size;
  const float* x  = (const float*)d_in[0];
  const float* Wc = (const float*)d_in[1];
  const float* bc = (const float*)d_in[2];
  const float* Wq = (const float*)d_in[3];
  const float* bq = (const float*)d_in[4];
  const float* qs = (const float*)d_in[5];
  float* out = (float*)d_out;

  bf16_t*   Wpk  = (bf16_t*)d_ws;
  bf16_t*   hbuf = (bf16_t*)((char*)d_ws + WPK_BYTES);
  unsigned* bar  = (unsigned*)((char*)d_ws + WPK_BYTES + HBUF_BYTES);

  // zero h state (h0 = 0) and barrier counter; graph-capturable
  hipMemsetAsync((char*)d_ws + WPK_BYTES, 0, HBUF_BYTES + 256, stream);

  qlstm_pack_w<<<2048, 32, 0, stream>>>(Wc, Wq, Wpk);
  qlstm_persistent<<<NBLK, 256, 0, stream>>>(x, bc, bq, qs, Wpk, hbuf, bar,
                                             out);
}
